// GumbelVectorQuantizer_74801150427611
// MI455X (gfx1250) — matrix operand order repaired and checked by reading, not compiled, not run on hardware
//
#include <hip/hip_runtime.h>
#include <hip/hip_bf16.h>
#include <stdint.h>

// ---------------------------------------------------------------------------
// GumbelVectorQuantizer for MI455X (gfx1250): WMMA f32<-f16 16x16x32 GEMMs
// with double-buffered LDS tiles fed by gfx1250 async global->LDS copies,
// exact JAX threefry2x32 Gumbel noise, LDS argmax reduction.
// ---------------------------------------------------------------------------

typedef __attribute__((ext_vector_type(16))) _Float16 v16h;
typedef __attribute__((ext_vector_type(8)))  _Float16 v8h;
typedef __attribute__((ext_vector_type(8)))  float    v8f;
typedef __attribute__((ext_vector_type(4)))  int      v4i;

union V16 { v16h v; v8h h[2]; };

#define BT      16384          // B*T
#define DDIM    1024
#define NGRP    16             // V
#define NCODE   512            // K
#define GDIM    64             // d = D/V

// ---------------------------------------------------------------------------
// gfx1250 async global->LDS copy (ASYNCcnt-tracked), with sync fallback.
// Builtin signature (from hipcc diagnostic): param0 = v4i addrspace(1)*
// (global source), param1 = LDS side, then imm offset / imm cpol.
// ---------------------------------------------------------------------------
#if defined(__gfx1250__) && __has_builtin(__builtin_amdgcn_global_load_async_to_lds_b128)
#define HAVE_ASYNC_LDS 1
#else
#define HAVE_ASYNC_LDS 0
#endif

typedef __attribute__((address_space(1))) v4i* g_v4i_ptr;
typedef __attribute__((address_space(3))) v4i* l_v4i_ptr;

__device__ __forceinline__ void async_copy16(const _Float16* gsrc, _Float16* lds_dst)
{
#if HAVE_ASYNC_LDS
    __builtin_amdgcn_global_load_async_to_lds_b128(
        (g_v4i_ptr)gsrc, (l_v4i_ptr)lds_dst, 0, 0);
#else
    *(v8h*)lds_dst = *(const v8h*)gsrc;
#endif
}

__device__ __forceinline__ void async_wait_all()
{
#if HAVE_ASYNC_LDS
#if __has_builtin(__builtin_amdgcn_s_wait_asynccnt)
    __builtin_amdgcn_s_wait_asynccnt(0);
#else
    asm volatile("s_wait_asynccnt 0x0" ::: "memory");
#endif
#endif
}

// ---------------------------------------------------------------------------
// f32 -> f16 conversion (4 elements / thread)
// ---------------------------------------------------------------------------
__global__ __launch_bounds__(256) void cvt_f32_to_f16(
    const float* __restrict__ in, _Float16* __restrict__ out, int n)
{
    int i = (blockIdx.x * 256 + threadIdx.x) * 4;
    if (i + 3 < n) {
        float4 f = *(const float4*)(in + i);
        out[i + 0] = (_Float16)f.x;
        out[i + 1] = (_Float16)f.y;
        out[i + 2] = (_Float16)f.z;
        out[i + 3] = (_Float16)f.w;
    }
}

// ---------------------------------------------------------------------------
// c2[v,k] = sum_j codebooks[v,k,j]^2   (f32 source for accuracy)
// ---------------------------------------------------------------------------
__global__ __launch_bounds__(256) void c2_kernel(
    const float* __restrict__ cb, float* __restrict__ c2)
{
    int g = blockIdx.x * 256 + threadIdx.x;      // (v*512 + k)
    if (g < NGRP * NCODE) {
        const float* p = cb + (size_t)g * GDIM;
        float s = 0.f;
        #pragma unroll 8
        for (int j = 0; j < GDIM; ++j) { float f = p[j]; s += f * f; }
        c2[g] = s;
    }
}

// ---------------------------------------------------------------------------
// q2[token,v] = sum_j q16[token, v*64+j]^2
// ---------------------------------------------------------------------------
__global__ __launch_bounds__(256) void q2_kernel(
    const _Float16* __restrict__ q16, float* __restrict__ q2)
{
    int g = blockIdx.x * 256 + threadIdx.x;      // token*16 + v
    if (g < BT * NGRP) {
        const _Float16* p = q16 + (size_t)(g >> 4) * DDIM + (g & 15) * GDIM;
        float s = 0.f;
        #pragma unroll 8
        for (int j = 0; j < GDIM; ++j) { float f = (float)p[j]; s += f * f; }
        q2[g] = s;
    }
}

// ---------------------------------------------------------------------------
// WMMA GEMM:  C[M,N] = A16[M,K] * W16[N,K]^T + bias
// block = 256 threads = 8 waves; block tile = 128(M) x 64(N).
// Wave w owns rows w*16..w*16+15 and all 64 columns: 4 accumulators.
// A and W tiles double-buffered in LDS (rows padded to 40 halves so the 16
// column addresses land in 16 distinct banks), filled by async global->LDS
// b128 copies that overlap the WMMA work of the current tile.
// ---------------------------------------------------------------------------
#define LPAD 40

__global__ __launch_bounds__(256) void gemm_wmma(
    const _Float16* __restrict__ A, const _Float16* __restrict__ W,
    const float* __restrict__ bias,
    float* __restrict__ Cf32, _Float16* __restrict__ Cf16,
    int M, int N, int K, int outF16)
{
    __shared__ __align__(16) _Float16 Ab[2][128][LPAD];   // 20 KB
    __shared__ __align__(16) _Float16 Bb[2][64][LPAD];    // 10 KB

    const int nBlk  = N >> 6;
    const int nb    = blockIdx.x % nBlk;
    const int mb    = blockIdx.x / nBlk;
    const int nbase = nb * 64;
    const int mbase = mb * 128;
    const int tid   = threadIdx.x;
    const int wave  = tid >> 5;
    const int lane  = tid & 31;
    const int col   = lane & 15;
    const int hi    = lane >> 4;

    // staging: A tile = 128x32 halves = 512 b128 chunks (2/thread),
    //          W tile =  64x32 halves = 256 b128 chunks (1/thread)
    const int ac0r = tid >> 2,          ac0c = (tid & 3) * 8;
    const int ac1r = (tid + 256) >> 2,  ac1c = ac0c;
    const int wcr  = tid >> 2,          wcc  = (tid & 3) * 8;

    v8f acc0 = {}, acc1 = {}, acc2 = {}, acc3 = {};

    // prologue: stage k-step 0 into buffer 0
    async_copy16(A + (size_t)(mbase + ac0r) * K + ac0c,      &Ab[0][ac0r][ac0c]);
    async_copy16(A + (size_t)(mbase + ac1r) * K + ac1c,      &Ab[0][ac1r][ac1c]);
    async_copy16(W + (size_t)(nbase + wcr)  * K + wcc,       &Bb[0][wcr][wcc]);
    async_wait_all();
    __syncthreads();

    int p = 0;
    for (int kb = 0; kb < K; kb += 32) {
        // stage next k-step into the other buffer while computing this one
        if (kb + 32 < K) {
            int kn = kb + 32;
            async_copy16(A + (size_t)(mbase + ac0r) * K + kn + ac0c, &Ab[p ^ 1][ac0r][ac0c]);
            async_copy16(A + (size_t)(mbase + ac1r) * K + kn + ac1c, &Ab[p ^ 1][ac1r][ac1c]);
            async_copy16(W + (size_t)(nbase + wcr)  * K + kn + wcc,  &Bb[p ^ 1][wcr][wcc]);
        }

        // A fragment (ISA 16-bit A layout: K = hi*8+{0..7}, 16+hi*8+{0..7})
        V16 a;
        a.h[0] = *(const v8h*)(&Ab[p][wave * 16 + col][hi * 8]);
        a.h[1] = *(const v8h*)(&Ab[p][wave * 16 + col][16 + hi * 8]);

        // 4 B fragments (K = hi*8+{0..7}, 16+hi*8+{0..7}, matching A)
        V16 b0, b1, b2, b3;
        b0.h[0] = *(const v8h*)(&Bb[p][col][hi * 8]);
        b0.h[1] = *(const v8h*)(&Bb[p][col][16 + hi * 8]);
        b1.h[0] = *(const v8h*)(&Bb[p][16 + col][hi * 8]);
        b1.h[1] = *(const v8h*)(&Bb[p][16 + col][16 + hi * 8]);
        b2.h[0] = *(const v8h*)(&Bb[p][32 + col][hi * 8]);
        b2.h[1] = *(const v8h*)(&Bb[p][32 + col][16 + hi * 8]);
        b3.h[0] = *(const v8h*)(&Bb[p][48 + col][hi * 8]);
        b3.h[1] = *(const v8h*)(&Bb[p][48 + col][16 + hi * 8]);

        acc0 = __builtin_amdgcn_wmma_f32_16x16x32_f16(false, a.v, false, b0.v, (short)0, acc0, false, false);
        acc1 = __builtin_amdgcn_wmma_f32_16x16x32_f16(false, a.v, false, b1.v, (short)0, acc1, false, false);
        acc2 = __builtin_amdgcn_wmma_f32_16x16x32_f16(false, a.v, false, b2.v, (short)0, acc2, false, false);
        acc3 = __builtin_amdgcn_wmma_f32_16x16x32_f16(false, a.v, false, b3.v, (short)0, acc3, false, false);

        async_wait_all();        // own async copies done; barrier makes all visible
        __syncthreads();
        p ^= 1;
    }

    // epilogue: C layout per tile: n = col, VGPR r -> m = r + 8*hi
    const int mrow = mbase + wave * 16 + 8 * hi;
    #pragma unroll
    for (int j = 0; j < 4; ++j) {
        const v8f* accp = (j == 0) ? &acc0 : (j == 1) ? &acc1 : (j == 2) ? &acc2 : &acc3;
        const int n = nbase + j * 16 + col;
        const float bval = bias[n];
        #pragma unroll
        for (int r = 0; r < 8; ++r) {
            float val = (*accp)[r] + bval;
            size_t o = (size_t)(mrow + r) * N + n;
            if (outF16) Cf16[o] = (_Float16)val;
            else        Cf32[o] = val;
        }
    }
}

// ---------------------------------------------------------------------------
// Exact JAX threefry2x32 (key = jax.random.key(42) -> [0,42]) Gumbel noise
// ---------------------------------------------------------------------------
__device__ __forceinline__ unsigned rotl32(unsigned x, int r)
{ return (x << r) | (x >> (32 - r)); }

__device__ __forceinline__ float gumbel_at(unsigned i)
{
    const unsigned HALF = 1u << 26;              // n = 2^27 elements
    unsigned x0, x1; bool second;
    if (i < HALF) { x0 = i;        x1 = i + HALF; second = false; }
    else          { x0 = i - HALF; x1 = i;        second = true;  }
    const unsigned ks0 = 0u, ks1 = 42u, ks2 = 0x1BD11BDAu ^ ks0 ^ ks1;
    x0 += ks0; x1 += ks1;
#define TF_R4(ra, rb, rc, rd, A0, A1)                         \
    x0 += x1; x1 = rotl32(x1, ra); x1 ^= x0;                  \
    x0 += x1; x1 = rotl32(x1, rb); x1 ^= x0;                  \
    x0 += x1; x1 = rotl32(x1, rc); x1 ^= x0;                  \
    x0 += x1; x1 = rotl32(x1, rd); x1 ^= x0;                  \
    x0 += (A0); x1 += (A1);
    TF_R4(13, 15, 26,  6, ks1, ks2 + 1u)
    TF_R4(17, 29, 16, 24, ks2, ks0 + 2u)
    TF_R4(13, 15, 26,  6, ks0, ks1 + 3u)
    TF_R4(17, 29, 16, 24, ks1, ks2 + 4u)
    TF_R4(13, 15, 26,  6, ks2, ks0 + 5u)
#undef TF_R4
    unsigned bits = second ? x1 : x0;
    float u = __uint_as_float((bits >> 9) | 0x3F800000u) - 1.0f;
    float t = -__logf(u + 1e-8f);
    return -__logf(t + 1e-8f);
}

// ---------------------------------------------------------------------------
// Distance + gumbel + argmax.  Block = (16-token tile, group v); 8 waves.
// Wave w covers code tiles nt in {w, w+8, w+16, w+24}; each tile is a
// 16x16 WMMA output C[code, token] = cb(16x64) . q(64x16), K-loop of 2.
// ---------------------------------------------------------------------------
__global__ __launch_bounds__(256) void argmax_kernel(
    const _Float16* __restrict__ q16, const _Float16* __restrict__ cb16,
    const float* __restrict__ q2, const float* __restrict__ c2,
    int* __restrict__ idx_out, int* __restrict__ counts)
{
    const int v    = blockIdx.x & 15;
    const int tt   = blockIdx.x >> 4;
    const int wave = threadIdx.x >> 5;
    const int lane = threadIdx.x & 31;
    const int col  = lane & 15;                  // token (B's N) / code row (A)
    const int hi   = lane >> 4;
    const int token = tt * 16 + col;

    const float q2v = q2[token * NGRP + v];

    // B fragments (q for this token/group) are nt-invariant: preload both k-steps
    const _Float16* qp = q16 + (size_t)token * DDIM + v * GDIM;
    V16 b0, b1;
    b0.h[0] = *(const v8h*)(qp + hi * 8);
    b0.h[1] = *(const v8h*)(qp + 16 + hi * 8);
    b1.h[0] = *(const v8h*)(qp + 32 + hi * 8);
    b1.h[1] = *(const v8h*)(qp + 48 + hi * 8);

    float best = -INFINITY; int bestk = 0;

    for (int ntl = 0; ntl < 4; ++ntl) {
        const int nt = wave + ntl * 8;
        const int codeBase = nt * 16;
        const _Float16* cp = cb16 + ((size_t)(v * NCODE + codeBase + col)) * GDIM;

        V16 a; v8f c = {};
        a.h[0] = *(const v8h*)(cp + hi * 8);
        a.h[1] = *(const v8h*)(cp + 16 + hi * 8);
        c = __builtin_amdgcn_wmma_f32_16x16x32_f16(
                false, a.v, false, b0.v, (short)0, c, false, false);
        a.h[0] = *(const v8h*)(cp + 32 + hi * 8);
        a.h[1] = *(const v8h*)(cp + 48 + hi * 8);
        c = __builtin_amdgcn_wmma_f32_16x16x32_f16(
                false, a.v, false, b1.v, (short)0, c, false, false);

        #pragma unroll
        for (int r = 0; r < 8; ++r) {
            int k = codeBase + r + 8 * hi;       // C: m = r + 8*hi (= code)
            float d2 = q2v + c2[v * NCODE + k] - 2.0f * c[r];
            float s  = -__fsqrt_rn(fmaxf(d2, 0.0f))
                     + gumbel_at(((unsigned)token * NGRP + v) * NCODE + k);
            if (s > best || (s == best && k < bestk)) { best = s; bestk = k; }
        }
    }

    __shared__ float sv[16][16];
    __shared__ int   sk[16][16];
    sv[col][wave * 2 + hi] = best;
    sk[col][wave * 2 + hi] = bestk;
    __syncthreads();

    if (threadIdx.x < 16) {
        int t = threadIdx.x;
        float bb = sv[t][0]; int bk = sk[t][0];
        #pragma unroll
        for (int s = 1; s < 16; ++s) {
            float vv = sv[t][s]; int kk = sk[t][s];
            if (vv > bb || (vv == bb && kk < bk)) { bb = vv; bk = kk; }
        }
        idx_out[(tt * 16 + t) * NGRP + v] = bk;
        atomicAdd(&counts[v * NCODE + bk], 1);
    }
}

// ---------------------------------------------------------------------------
// targets[token] = sum_v idx[token,v] * (v*K)    (int32)
// ---------------------------------------------------------------------------
__global__ __launch_bounds__(256) void targets_kernel(
    const int* __restrict__ idx, int* __restrict__ targets)
{
    int t = blockIdx.x * 256 + threadIdx.x;
    if (t < BT) {
        int s = 0;
        #pragma unroll
        for (int v = 0; v < NGRP; ++v) s += idx[t * NGRP + v] * (v * NCODE);
        targets[t] = s;
    }
}

// ---------------------------------------------------------------------------
// Gather selected codewords: qvars16[token, v*64+j] = cb16[v, idx, j]
// ---------------------------------------------------------------------------
__global__ __launch_bounds__(256) void gather_kernel(
    const _Float16* __restrict__ cb16, const int* __restrict__ idx,
    _Float16* __restrict__ qvars)
{
    int g = blockIdx.x * 256 + threadIdx.x;      // token*128 + v*8 + jblk
    if (g < BT * NGRP * (GDIM / 8)) {
        int jblk  = g & 7;
        int v     = (g >> 3) & 15;
        int token = g >> 7;
        int k = idx[token * NGRP + v];
        v8h val = *(const v8h*)(cb16 + ((size_t)(v * NCODE + k)) * GDIM + jblk * 8);
        *(v8h*)(qvars + (size_t)token * DDIM + v * GDIM + jblk * 8) = val;
    }
}

// ---------------------------------------------------------------------------
// Diversity loss.  Per-group total is exactly B*T = 16384.
// ---------------------------------------------------------------------------
__global__ __launch_bounds__(256) void losses_kernel(
    const int* __restrict__ counts, float* __restrict__ out_loss)
{
    __shared__ float part[256];
    float s = 0.f;
    for (int i = threadIdx.x; i < NGRP * NCODE; i += 256) {
        float p = (float)counts[i] * (1.0f / 16384.0f);
        s += -p * __logf(p + 1e-8f);
    }
    part[threadIdx.x] = s;
    __syncthreads();
    for (int st = 128; st > 0; st >>= 1) {
        if (threadIdx.x < st) part[threadIdx.x] += part[threadIdx.x + st];
        __syncthreads();
    }
    if (threadIdx.x == 0) {
        float diversity = -(part[0] / (float)NGRP) / __logf((float)NCODE);
        out_loss[0] = 0.1f * diversity;
    }
}

// ---------------------------------------------------------------------------
// Host launcher
// ---------------------------------------------------------------------------
extern "C" void kernel_launch(void* const* d_in, const int* in_sizes, int n_in,
                              void* d_out, int out_size, void* d_ws, size_t ws_size,
                              hipStream_t stream)
{
    const float* features  = (const float*)d_in[0];  // (8,2048,1024)
    const float* codebooks = (const float*)d_in[1];  // (16,512,64)
    const float* Wq   = (const float*)d_in[2];       // (1024,1024)
    const float* bq   = (const float*)d_in[3];       // (1024)
    const float* Wout = (const float*)d_in[4];       // (1024,1024)
    const float* bout = (const float*)d_in[5];       // (1024)

    char* ws = (char*)d_ws;
    _Float16* feat16  = (_Float16*)(ws);                         // 32 MB (reused as qvars16)
    _Float16* q16     = (_Float16*)(ws + 33554432);              // 32 MB
    _Float16* Wq16    = (_Float16*)(ws + 67108864);              // 2 MB
    _Float16* Wout16  = (_Float16*)(ws + 69206016);              // 2 MB
    _Float16* cb16    = (_Float16*)(ws + 71303168);              // 1 MB
    float*    c2buf   = (float*)   (ws + 72351744);              // 32 KB
    float*    q2buf   = (float*)   (ws + 72384512);              // 1 MB
    int*      idxbuf  = (int*)     (ws + 73433088);              // 1 MB
    int*      counts  = (int*)     (ws + 74481664);              // 32 KB

    float* out_quant  = (float*)d_out;                           // 16777216 f32
    int*   out_tgt    = (int*)d_out + (BT * DDIM);               // 16384 int32
    float* out_loss   = (float*)d_out + (BT * DDIM) + BT;        // 1 f32

    // --- precision conversions
    cvt_f32_to_f16<<<(BT * DDIM) / 1024, 256, 0, stream>>>(features, feat16, BT * DDIM);
    cvt_f32_to_f16<<<(DDIM * DDIM) / 1024, 256, 0, stream>>>(Wq, Wq16, DDIM * DDIM);
    cvt_f32_to_f16<<<(DDIM * DDIM) / 1024, 256, 0, stream>>>(Wout, Wout16, DDIM * DDIM);
    cvt_f32_to_f16<<<(NGRP * NCODE * GDIM) / 1024, 256, 0, stream>>>(
        codebooks, cb16, NGRP * NCODE * GDIM);

    // --- codebook squared norms (from f32 source)
    c2_kernel<<<(NGRP * NCODE + 255) / 256, 256, 0, stream>>>(codebooks, c2buf);

    (void)hipMemsetAsync(counts, 0, NGRP * NCODE * sizeof(int), stream);

    // --- GEMM1: q = features @ Wq^T + bq  (f16 output for the distance stage)
    gemm_wmma<<<(BT / 128) * (DDIM / 64), 256, 0, stream>>>(
        feat16, Wq16, bq, nullptr, q16, BT, DDIM, DDIM, 1);

    // --- per-(token,group) squared norms
    q2_kernel<<<(BT * NGRP) / 256, 256, 0, stream>>>(q16, q2buf);

    // --- distances + threefry gumbel + argmax (+ histogram)
    argmax_kernel<<<(BT / 16) * NGRP, 256, 0, stream>>>(
        q16, cb16, q2buf, c2buf, idxbuf, counts);

    // --- contrastive targets
    targets_kernel<<<BT / 256, 256, 0, stream>>>(idxbuf, out_tgt);

    // --- codebook gather (reuses feat16 region as qvars16)
    gather_kernel<<<(BT * NGRP * (GDIM / 8)) / 256, 256, 0, stream>>>(
        cb16, idxbuf, feat16);

    // --- GEMM2: quantized = qvars @ Wout^T + bout  (f32 to d_out)
    gemm_wmma<<<(BT / 128) * (DDIM / 64), 256, 0, stream>>>(
        feat16, Wout16, bout, out_quant, nullptr, BT, DDIM, DDIM, 0);

    // --- diversity loss
    losses_kernel<<<1, 256, 0, stream>>>(counts, out_loss);
}
